// NaiveMultiHeadSelfAttention_48249662603266
// MI455X (gfx1250) — matrix operand order repaired and checked by reading, not compiled, not run on hardware
//
#include <hip/hip_runtime.h>

// ---------------------------------------------------------------------------
// Problem constants
// ---------------------------------------------------------------------------
#define BATCH   2
#define SEQ     2048
#define DIN     1024
#define NHEAD   16
#define DH      64          // head dim
#define DOUT    1024        // NHEAD * DH

typedef __attribute__((ext_vector_type(16))) __bf16 v16bf;   // 8 VGPRs
typedef __attribute__((ext_vector_type(8)))  float  v8f;     // 8 VGPRs
typedef __attribute__((ext_vector_type(2)))  float  vf2;
typedef __attribute__((ext_vector_type(2)))  __bf16 vbf2;

// fp32 pair -> packed bf16 (native v_cvt_pk_bf16_f32)
__device__ __forceinline__ unsigned pack2bf(float a, float b) {
  vf2 f = {a, b};
  vbf2 h = __builtin_convertvector(f, vbf2);
  return __builtin_bit_cast(unsigned, h);
}
__device__ __forceinline__ unsigned short f2bf1(float a) {
  __bf16 h = (__bf16)a;
  return __builtin_bit_cast(unsigned short, h);
}

union ABReg { v16bf bf; unsigned u[8]; uint4 q[2]; };

// ---------------------------------------------------------------------------
// Kernel 0: bulk fp32 -> bf16 conversion (weight matrices).
// ---------------------------------------------------------------------------
__global__ __launch_bounds__(256) void cvt_bf16_kernel(
    const float* __restrict__ src, unsigned* __restrict__ dst, int npairs)
{
  int i = blockIdx.x * blockDim.x + threadIdx.x;
  if (i < npairs) {
    float2 f = ((const float2*)src)[i];
    dst[i] = pack2bf(f.x, f.y);
  }
}

// ---------------------------------------------------------------------------
// Kernel 1: OUT = (X @ W^T + bias) * oscale, stored as bf16.
//   X:   [4096][1024] fp32     Wb: [1024][1024] bf16 (pre-converted)
//   vtranspose == 0 : OUT[((b*NHEAD+h)*SEQ + s)*DH + d]   (Q, K layout)
//   vtranspose == 1 : OUT[((b*NHEAD+h)*DH + d)*SEQ + s]   (V^T layout)
// One wave computes a 32x64 output tile (2 A tiles share each B tile).
// grid = (32, 16), block = 128 (4 waves).
// ---------------------------------------------------------------------------
__global__ __launch_bounds__(128) void qkv_proj_kernel(
    const float* __restrict__ X, const unsigned short* __restrict__ Wb,
    const float* __restrict__ bias, unsigned short* __restrict__ OUT,
    int vtranspose, float oscale)
{
  const int lane = threadIdx.x & 31;
  const int wave = threadIdx.x >> 5;
  const int half = lane >> 4;          // 0: lanes 0-15, 1: lanes 16-31
  const int l15  = lane & 15;

  const int m0 = (blockIdx.x * 4 + wave) * 32;   // row tile base (32 rows)
  const int n0 = blockIdx.y * 64;                // col tile base (64 cols)

  v8f z = {0.f,0.f,0.f,0.f,0.f,0.f,0.f,0.f};
  v8f acc0[4] = {z, z, z, z};
  v8f acc1[4] = {z, z, z, z};

  // loop-invariant row pointers (kc folds into instruction offsets)
  const float* Xr0 = X + (size_t)(m0 + l15) * DIN;
  const float* Xr1 = Xr0 + (size_t)16 * DIN;
  const unsigned short* Wp0 = Wb + (size_t)(n0 +  0 + l15) * DIN + half * 16;
  const unsigned short* Wp1 = Wb + (size_t)(n0 + 16 + l15) * DIN + half * 16;
  const unsigned short* Wp2 = Wb + (size_t)(n0 + 32 + l15) * DIN + half * 16;
  const unsigned short* Wp3 = Wb + (size_t)(n0 + 48 + l15) * DIN + half * 16;

  for (int kc = 0; kc < DIN; kc += 32) {
    // --- two A tiles (16x32 bf16 each): 8 x b128 loads + packed cvt ---
    ABReg A0, A1;
#pragma unroll
    for (int g = 0; g < 2; ++g) {
      const float* xp0 = Xr0 + kc + half * 16 + g * 8;
      const float* xp1 = Xr1 + kc + half * 16 + g * 8;
      float4 a0 = *(const float4*)(xp0);
      float4 a1 = *(const float4*)(xp0 + 4);
      float4 b0 = *(const float4*)(xp1);
      float4 b1 = *(const float4*)(xp1 + 4);
      A0.u[4*g+0] = pack2bf(a0.x, a0.y);  A0.u[4*g+1] = pack2bf(a0.z, a0.w);
      A0.u[4*g+2] = pack2bf(a1.x, a1.y);  A0.u[4*g+3] = pack2bf(a1.z, a1.w);
      A1.u[4*g+0] = pack2bf(b0.x, b0.y);  A1.u[4*g+1] = pack2bf(b0.z, b0.w);
      A1.u[4*g+2] = pack2bf(b1.x, b1.y);  A1.u[4*g+3] = pack2bf(b1.z, b1.w);
    }
    // --- 4 B tiles, each reused by both A tiles: 8 WMMAs per kc step ---
    const unsigned short* wp[4] = {Wp0, Wp1, Wp2, Wp3};
#pragma unroll
    for (int nt = 0; nt < 4; ++nt) {
      ABReg Bm;
      Bm.q[0] = *(const uint4*)(wp[nt] + kc);
      Bm.q[1] = *(const uint4*)(wp[nt] + kc + 8);
      acc0[nt] = __builtin_amdgcn_wmma_f32_16x16x32_bf16(
          false, A0.bf, false, Bm.bf, (short)0, acc0[nt], false, false);
      acc1[nt] = __builtin_amdgcn_wmma_f32_16x16x32_bf16(
          false, A1.bf, false, Bm.bf, (short)0, acc1[nt], false, false);
    }
  }

  // --- bias, scale, bf16 store (both 16-row groups) ---
#pragma unroll
  for (int nt = 0; nt < 4; ++nt) {
    const int n = n0 + nt * 16 + l15;
    const float bval = bias[n];
    const int h = n >> 6, d = n & (DH - 1);
#pragma unroll
    for (int grp = 0; grp < 2; ++grp) {
      const v8f* acc = grp ? acc1 : acc0;
#pragma unroll
      for (int r = 0; r < 8; ++r) {
        const int m = m0 + grp * 16 + (half << 3) + r;  // C layout M = r+8*half
        const int b = m >> 11, s = m & (SEQ - 1);
        const unsigned short o = f2bf1((acc[nt][r] + bval) * oscale);
        size_t idx;
        if (vtranspose)
          idx = ((size_t)(b * NHEAD + h) * DH + d) * SEQ + s;
        else
          idx = ((size_t)(b * NHEAD + h) * SEQ + s) * DH + d;
        OUT[idx] = o;
      }
    }
  }
}

// ---------------------------------------------------------------------------
// Kernel 2: flash-style attention per (b,h) head, transposed-score scheme.
//   Qb (pre-scaled by 1/sqrt(DH)), Kb: bf16 [B*NHEAD][SEQ][DH]
//   Vt: bf16 [B*NHEAD][DH][SEQ]
//   out: fp32 [B][SEQ][NHEAD*DH]
//
// Scores computed transposed: St = K-tile(A) x Q-tile(B)  ->  C layout has
// query = lane (N), key = VGPR row (M = r + 8*half).  That is exactly the
// per-lane data the A-operand of P@V needs, so P is re-packed to bf16
// IN-LANE (no LDS, no shuffles).  Softmax row stats are per-lane scalars;
// only the O-rescale needs an 8-way query-indexed broadcast.
// One wave owns a 16-query tile, keys swept in blocks of 32.
// grid = (32, 32), block = 128 (4 independent waves).
// ---------------------------------------------------------------------------
__global__ __launch_bounds__(128) void attn_kernel(
    const unsigned short* __restrict__ Qb,
    const unsigned short* __restrict__ Kb,
    const unsigned short* __restrict__ Vt,
    float* __restrict__ out)
{
  const int lane = threadIdx.x & 31;
  const int wave = threadIdx.x >> 5;
  const int half = lane >> 4;
  const int l15  = lane & 15;

  const int bh = blockIdx.x;                     // 0..31
  const int b  = bh >> 4, h = bh & (NHEAD - 1);
  const int q0 = (blockIdx.y * 4 + wave) * 16;   // query tile base

  const unsigned short* Qh = Qb + (size_t)bh * SEQ * DH;
  const unsigned short* Kh = Kb + (size_t)bh * SEQ * DH;
  const unsigned short* Vh = Vt + (size_t)bh * DH * SEQ;

  // --- load Q tile once as the B-operand of St = K x Q^T ---
  // B layout: lane = query column (l15); vgpr i holds d = c*32 + 16*half + 2i
  ABReg Qm[2];
  {
    const unsigned short* qp = Qh + (size_t)(q0 + l15) * DH + half * 16;
#pragma unroll
    for (int c = 0; c < 2; ++c) {
      Qm[c].q[0] = *(const uint4*)(qp + c * 32);
      Qm[c].q[1] = *(const uint4*)(qp + c * 32 + 8);
    }
  }

  v8f z = {0.f,0.f,0.f,0.f,0.f,0.f,0.f,0.f};
  v8f O[4] = {z, z, z, z};
  float mrun = -1e30f, lrun = 0.f;   // per-lane scalars: this lane's query l15

  // running pointers: all loads below use constant immediate offsets
  // K A-operand: lane = key row (l15); vgpr i: d = 16*half + 2i
  const unsigned short* Krun = Kh + (size_t)l15 * DH + half * 16;  // += 32*DH
  const unsigned short* Vrun = Vh + (size_t)l15 * SEQ + half * 8;  // += 32

  for (int kb = 0; kb < SEQ; kb += 32, Krun += 32 * DH, Vrun += 32) {
    if (kb + 32 < SEQ) {   // uniform branch: prefetch next key/value block
      __builtin_prefetch(Krun + 32 * DH + (lane & 1) * 16 * DH, 0, 0);
      __builtin_prefetch(Vrun + 32 + (lane & 3) * 16 * SEQ, 0, 0);
    }

    // --- St[key, query] for keys kb+t*16.. : 4 WMMAs (Q pre-scaled) ---
    v8f Sc[2] = {z, z};
#pragma unroll
    for (int t = 0; t < 2; ++t) {
#pragma unroll
      for (int c = 0; c < 2; ++c) {
        ABReg Ak;   // K tile as A operand
        const uint4* kp = (const uint4*)(Krun + t * 16 * DH + c * 32);
        Ak.q[0] = kp[0];                    // d = c*32 + 16*half + 0..7
        Ak.q[1] = kp[1];                    // d = c*32 + 16*half + 8..15
        Sc[t] = __builtin_amdgcn_wmma_f32_16x16x32_bf16(
            false, Ak.bf, false, Qm[c].bf, (short)0, Sc[t], false, false);
      }
    }

    // --- online softmax: per-lane scalar stats for query l15 ---
    float v = fmaxf(Sc[0][0], Sc[1][0]);
#pragma unroll
    for (int r = 1; r < 8; ++r) v = fmaxf(v, fmaxf(Sc[0][r], Sc[1][r]));
    v = fmaxf(v, __shfl_xor(v, 16, 32));           // other half's key subset
    const float mnew = fmaxf(mrun, v);
    const float scalef = __expf(mrun - mnew);
    mrun = mnew;

    float s = 0.f;
#pragma unroll
    for (int r = 0; r < 8; ++r) {
      Sc[0][r] = __expf(Sc[0][r] - mnew);
      Sc[1][r] = __expf(Sc[1][r] - mnew);
      s += Sc[0][r] + Sc[1][r];
    }
    s += __shfl_xor(s, 16, 32);
    lrun = lrun * scalef + s;

    // --- broadcast per-query rescale factors to the O layout (row = r+8*half)
#pragma unroll
    for (int dt = 0; dt < 4; ++dt)
#pragma unroll
      for (int r = 0; r < 8; ++r)
        O[dt][r] *= __shfl(scalef, (half << 3) + r, 32);

    // --- P: in-lane repack of St (already the A-operand arrangement) ---
    ABReg Pa;
#pragma unroll
    for (int i = 0; i < 4; ++i) {
      Pa.u[i]     = pack2bf(Sc[0][2*i], Sc[0][2*i+1]);  // keys 8*half + 2i,+1
      Pa.u[4 + i] = pack2bf(Sc[1][2*i], Sc[1][2*i+1]);  // keys 16+8*half+2i,+1
    }

    // --- O += P @ V : 4 WMMAs over the 4 d-tiles (V stored transposed) ---
#pragma unroll
    for (int dt = 0; dt < 4; ++dt) {
      ABReg Bv;
      const uint4* vp = (const uint4*)(Vrun + (size_t)dt * 16 * SEQ);
      Bv.q[0] = vp[0];
      Bv.q[1] = vp[2];
      O[dt] = __builtin_amdgcn_wmma_f32_16x16x32_bf16(
          false, Pa.bf, false, Bv.bf, (short)0, O[dt], false, false);
    }
  }

  // --- epilogue: broadcast 1/l per query row, fp32 stores ---
  const float linv = 1.0f / lrun;
#pragma unroll
  for (int r = 0; r < 8; ++r) {
    const int srow = q0 + (half << 3) + r;
    const float inv = __shfl(linv, (half << 3) + r, 32);
    float* orow = out + ((size_t)b * SEQ + srow) * DOUT + h * DH;
#pragma unroll
    for (int dt = 0; dt < 4; ++dt)
      orow[dt * 16 + l15] = O[dt][r] * inv;
  }
}

// ---------------------------------------------------------------------------
// Launch: convert weights to bf16, 3 projection passes, attention.
// Workspace: Q | K | V^T (8MB each, bf16) then Wq|Wk|Wv (2MB each, bf16).
// ---------------------------------------------------------------------------
extern "C" void kernel_launch(void* const* d_in, const int* in_sizes, int n_in,
                              void* d_out, int out_size, void* d_ws, size_t ws_size,
                              hipStream_t stream) {
  const float* xq = (const float*)d_in[0];
  const float* xk = (const float*)d_in[1];
  const float* xv = (const float*)d_in[2];
  const float* wq = (const float*)d_in[3];
  const float* bq = (const float*)d_in[4];
  const float* wk = (const float*)d_in[5];
  const float* bk = (const float*)d_in[6];
  const float* wv = (const float*)d_in[7];
  const float* bv = (const float*)d_in[8];
  float* out = (float*)d_out;

  const size_t qkv_elems = (size_t)BATCH * SEQ * DOUT;   // 4,194,304
  const size_t w_elems   = (size_t)DOUT * DIN;           // 1,048,576
  unsigned short* Qb  = (unsigned short*)d_ws;
  unsigned short* Kb  = Qb + qkv_elems;
  unsigned short* Vt  = Kb + qkv_elems;
  unsigned short* Wqb = Vt + qkv_elems;
  unsigned short* Wkb = Wqb + w_elems;
  unsigned short* Wvb = Wkb + w_elems;

  const int npairs = (int)(w_elems / 2);                 // 524,288
  dim3 cgrid((npairs + 255) / 256), cblock(256);
  cvt_bf16_kernel<<<cgrid, cblock, 0, stream>>>(wq, (unsigned*)Wqb, npairs);
  cvt_bf16_kernel<<<cgrid, cblock, 0, stream>>>(wk, (unsigned*)Wkb, npairs);
  cvt_bf16_kernel<<<cgrid, cblock, 0, stream>>>(wv, (unsigned*)Wvb, npairs);

  const float inv_temp = 0.125f;                         // 1/sqrt(DH)
  dim3 pgrid(32, 16), pblock(128);
  qkv_proj_kernel<<<pgrid, pblock, 0, stream>>>(xq, Wqb, bq, Qb, 0, inv_temp);
  qkv_proj_kernel<<<pgrid, pblock, 0, stream>>>(xk, Wkb, bk, Kb, 0, 1.0f);
  qkv_proj_kernel<<<pgrid, pblock, 0, stream>>>(xv, Wvb, bv, Vt, 1, 1.0f);

  dim3 agrid(BATCH * NHEAD, SEQ / 64), ablock(128);
  attn_kernel<<<agrid, ablock, 0, stream>>>(Qb, Kb, Vt, out);
}
